// GlobalGRUDecoder_15925738733984
// MI455X (gfx1250) — matrix operand order repaired and checked by reading, not compiled, not run on hardware
//
#include <hip/hip_runtime.h>

// ---------------------------------------------------------------------------
// GlobalGRUDecoder for MI455X (gfx1250, wave32, WMMA bf16 16x16x32)
//   V=10000 ENC=2048 D=512 E=512 B=128 L=64
// All GEMMs run through v_wmma_f32_16x16x32_bf16 with f32 accumulation.
// Block tile 64x128, wave tile 32x32 (4 wmma / K-slab), guard-free staging
// (V padded to 10112), double-buffered async global->LDS staging:
// slab k+1 DMA (GLOBAL_LOAD_ASYNC_TO_LDS_B128) overlaps slab k WMMAs via
// s_wait_asynccnt threshold waits (async loads retire in order).
// ---------------------------------------------------------------------------

#define CFG_V    10000
#define CFG_VP   10112        // padded vocab: 79 * 128
#define CFG_ENC  2048
#define CFG_D    512
#define CFG_E    512
#define CFG_B    128
#define CFG_L    64
#define CFG_L1   63           // L-1 decode steps
#define CFG_G    1536         // 3*D gate width

typedef __attribute__((ext_vector_type(16))) __bf16 v16bf;
typedef __attribute__((ext_vector_type(8)))  __bf16 v8bf;
typedef __attribute__((ext_vector_type(8)))  float  v8f;
typedef __attribute__((ext_vector_type(4)))  int    v4i;

union FragBF { v16bf v; v8bf h[2]; };

__device__ __forceinline__ unsigned short f2bf(float f) {
    unsigned int u = __float_as_uint(f);
    u += 0x7FFFu + ((u >> 16) & 1u);     // round-to-nearest-even
    return (unsigned short)(u >> 16);
}

__device__ __forceinline__ float sigmoidf_(float x) {
    return 1.0f / (1.0f + __expf(-x));
}

// ---- CDNA5 async global->LDS staging (ASYNCcnt path), with fallback --------
#if defined(__gfx1250__) && \
    __has_builtin(__builtin_amdgcn_global_load_async_to_lds_b128) && \
    __has_builtin(__builtin_amdgcn_s_wait_asynccnt)
#define GRU_ASYNC_LDS 1
#endif

#ifdef GRU_ASYNC_LDS
__device__ __forceinline__ void async_copy_b128(const void* g, void* lds) {
    // builtin expects (global int4*, local int4*, imm offset, imm cpol)
    __builtin_amdgcn_global_load_async_to_lds_b128(
        (__attribute__((address_space(1))) v4i*)(unsigned long long)g,
        (__attribute__((address_space(3))) v4i*)lds,
        0, 0);
}
#endif

// ---------------------------------------------------------------------------
// Tiled WMMA GEMM:  C[M,N] = act( A[M,K] @ W[N,K]^T + bias[N] )
//   Requirements: M % 64 == 0, N % 128 == 0, K % 32 == 0 (caller guarantees,
//   padding W for the vocab projection). ldc = real output width (<= N);
//   columns >= ldc are computed but not stored.
//   256 threads = 8 waves; block tile 64(M) x 128(N); wave tile 32x32.
//   perm: 0 = store row m
//         1 = m = b*63+l  -> store row l*128+b   (gi:  blg -> lbg)
//         2 = m = l*128+b -> store row b*63+l    (logits: lbv -> blv)
//   act : 0 = none, 1 = tanh
//   Cb  : optional bf16 mirror of C (same layout), may be null.
// ---------------------------------------------------------------------------
__global__ __launch_bounds__(256)
void gemm_bf16_wmma(const __bf16* __restrict__ A, const __bf16* __restrict__ W,
                    const float* __restrict__ bias,
                    float* __restrict__ C, unsigned short* __restrict__ Cb,
                    int M, int N, int K, int ldc, int act, int perm)
{
    __shared__ __align__(16) __bf16 Asl[2][64 * 32];    // 2 x 4 KB
    __shared__ __align__(16) __bf16 Bsl[2][128 * 32];   // 2 x 8 KB

    const int tid  = threadIdx.x;
    const int lane = tid & 31;
    const int wave = tid >> 5;          // 0..7
    const int wm   = wave >> 2;         // 0..1 : 32-row chunk of block
    const int wn   = wave & 3;          // 0..3 : 32-col chunk of block
    const int half = lane >> 4;         // 0: lanes 0-15, 1: lanes 16-31
    const int r15  = lane & 15;

    const int m0 = blockIdx.y * 64;
    const int n0 = blockIdx.x * 128;

    // staging coordinates (guard-free: dims are exact)
    const int arow = tid >> 2;          // 0..63
    const int acol = (tid & 3) * 8;     // 0,8,16,24
    const int brow = tid >> 1;          // 0..127
    const int bcol = (tid & 1) * 16;    // 0,16

    const __bf16* gaBase = A + (size_t)(m0 + arow) * K + acol;
    const __bf16* gbBase = W + (size_t)(n0 + brow) * K + bcol;

    // issue one slab's 3 b128 transfers into LDS buffer `buf`
    auto stage = [&](int k, int buf) {
        const __bf16* ga = gaBase + k;
        const __bf16* gb = gbBase + k;
#ifdef GRU_ASYNC_LDS
        async_copy_b128(ga,     &Asl[buf][arow * 32 + acol]);
        async_copy_b128(gb,     &Bsl[buf][brow * 32 + bcol]);
        async_copy_b128(gb + 8, &Bsl[buf][brow * 32 + bcol + 8]);
#else
        *reinterpret_cast<uint4*>(&Asl[buf][arow * 32 + acol]) =
            *reinterpret_cast<const uint4*>(ga);
        *reinterpret_cast<uint4*>(&Bsl[buf][brow * 32 + bcol]) =
            *reinterpret_cast<const uint4*>(gb);
        *reinterpret_cast<uint4*>(&Bsl[buf][brow * 32 + bcol + 8]) =
            *reinterpret_cast<const uint4*>(gb + 8);
#endif
    };

    v8f acc00 = {}, acc01 = {}, acc10 = {}, acc11 = {};

    const int nslabs = K >> 5;
    stage(0, 0);                                   // prologue: slab 0 -> buf 0

    for (int i = 0; i < nslabs; ++i) {
        const int cur = i & 1;
        if (i + 1 < nslabs) {
            stage((i + 1) << 5, cur ^ 1);          // prefetch next slab
#ifdef GRU_ASYNC_LDS
            // 3 newest (prefetch) may stay in flight; 3 older (slab i) retire
            // first because async loads complete in order.
            __builtin_amdgcn_s_wait_asynccnt(3);
#endif
        } else {
#ifdef GRU_ASYNC_LDS
            __builtin_amdgcn_s_wait_asynccnt(0);
#endif
        }
        __syncthreads();   // slab i visible to all waves

        // A fragments (16x32 bf16), documented CDNA5 lane layout:
        //  lanes 0-15 : halves 0-7 = K 0-7,  halves 8-15 = K 16-23
        //  lanes 16-31: halves 0-7 = K 8-15, halves 8-15 = K 24-31
        FragBF fa0, fa1, fb0, fb1;
        {
            const __bf16* ap0 = &Asl[cur][(wm * 32 + r15) * 32];
            const __bf16* ap1 = ap0 + 16 * 32;
            fa0.h[0] = *reinterpret_cast<const v8bf*>(ap0 + (half ? 8  : 0));
            fa0.h[1] = *reinterpret_cast<const v8bf*>(ap0 + (half ? 24 : 16));
            fa1.h[0] = *reinterpret_cast<const v8bf*>(ap1 + (half ? 8  : 0));
            fa1.h[1] = *reinterpret_cast<const v8bf*>(ap1 + (half ? 24 : 16));
        }
        // B fragments (32x16 bf16): lane = N column; lanes 0-15 hold K 0-7,16-23,
        // lanes 16-31 hold K 8-15,24-31 (same order as A)
        {
            const __bf16* bp0 = &Bsl[cur][(wn * 32 + r15) * 32];
            const __bf16* bp1 = bp0 + 16 * 32;
            const int kb = half ? 8 : 0;
            fb0.h[0] = *reinterpret_cast<const v8bf*>(bp0 + kb);
            fb0.h[1] = *reinterpret_cast<const v8bf*>(bp0 + kb + 16);
            fb1.h[0] = *reinterpret_cast<const v8bf*>(bp1 + kb);
            fb1.h[1] = *reinterpret_cast<const v8bf*>(bp1 + kb + 16);
        }

        acc00 = __builtin_amdgcn_wmma_f32_16x16x32_bf16(
                    false, fa0.v, false, fb0.v, (short)0, acc00, false, false);
        acc01 = __builtin_amdgcn_wmma_f32_16x16x32_bf16(
                    false, fa0.v, false, fb1.v, (short)0, acc01, false, false);
        acc10 = __builtin_amdgcn_wmma_f32_16x16x32_bf16(
                    false, fa1.v, false, fb0.v, (short)0, acc10, false, false);
        acc11 = __builtin_amdgcn_wmma_f32_16x16x32_bf16(
                    false, fa1.v, false, fb1.v, (short)0, acc11, false, false);
        __syncthreads();   // all waves done reading buf[cur] before it is
                           // overwritten by the prefetch issued next iteration
    }

    // ---- store: lane holds N = n, reg r holds M = r (+8 for upper half) ----
    v8f* accs[2][2] = { { &acc00, &acc01 }, { &acc10, &acc11 } };
    #pragma unroll
    for (int ni = 0; ni < 2; ++ni) {
        const int n = n0 + wn * 32 + ni * 16 + r15;
        if (n >= ldc) continue;
        const float bv = bias ? bias[n] : 0.0f;
        #pragma unroll
        for (int mi = 0; mi < 2; ++mi) {
            const v8f& a = *accs[mi][ni];
            #pragma unroll
            for (int r = 0; r < 8; ++r) {
                const int m = m0 + wm * 32 + mi * 16 + r + (half ? 8 : 0);
                float v = a[r] + bv;
                if (act == 1) v = tanhf(v);
                size_t off;
                if (perm == 0) {
                    off = (size_t)m * ldc + n;
                } else if (perm == 1) {       // m = b*63+l -> row l*128+b
                    const int b = m / CFG_L1, l = m % CFG_L1;
                    off = (size_t)(l * CFG_B + b) * ldc + n;
                } else {                       // m = l*128+b -> row b*63+l
                    const int l = m >> 7, b = m & (CFG_B - 1);
                    off = (size_t)(b * CFG_L1 + l) * ldc + n;
                }
                C[off] = v;
                if (Cb) Cb[off] = f2bf(v);
            }
        }
    }
}

// ---------------------------------------------------------------------------
// fp32 -> bf16 (RNE) elementwise converter, with optional row padding:
// rows >= rows_real are written as zero.
// ---------------------------------------------------------------------------
__global__ __launch_bounds__(256)
void f32_to_bf16_pad(const float* __restrict__ in,
                     unsigned short* __restrict__ out,
                     int rows_real, int cols, long long total)
{
    long long i = (long long)blockIdx.x * 256 + threadIdx.x;
    if (i >= total) return;
    const long long row = i / cols;
    out[i] = (row < rows_real) ? f2bf(in[i]) : (unsigned short)0;
}

// ---------------------------------------------------------------------------
// Embedding gather + convert: emb[b][l][e] = bf16(W_embed[cap[b][l]][e])
// ---------------------------------------------------------------------------
__global__ __launch_bounds__(256)
void embed_gather(const float* __restrict__ W_embed,
                  const long long* __restrict__ caps,
                  unsigned short* __restrict__ emb)
{
    int i = blockIdx.x * 256 + threadIdx.x;          // over B*L1*E
    if (i >= CFG_B * CFG_L1 * CFG_E) return;
    const int e   = i & (CFG_E - 1);
    const int row = i >> 9;                          // b*63 + l
    const int b   = row / CFG_L1;
    const int l   = row % CFG_L1;
    const long long tok = caps[(size_t)b * CFG_L + l];
    emb[i] = f2bf(W_embed[(size_t)tok * CFG_E + e]);
}

// ---------------------------------------------------------------------------
// GRU gate update for one timestep.
// ---------------------------------------------------------------------------
__global__ __launch_bounds__(256)
void gru_gate(const float* __restrict__ gi_t, const float* __restrict__ gh,
              float* __restrict__ h, unsigned short* __restrict__ h_bf,
              unsigned short* __restrict__ hs_t)
{
    int i = blockIdx.x * 256 + threadIdx.x;          // over B*D
    if (i >= CFG_B * CFG_D) return;
    const int b = i >> 9;
    const int d = i & (CFG_D - 1);
    const float* gir = gi_t + (size_t)b * CFG_G;
    const float* ghr = gh   + (size_t)b * CFG_G;

    const float r = sigmoidf_(gir[d]            + ghr[d]);
    const float z = sigmoidf_(gir[CFG_D + d]    + ghr[CFG_D + d]);
    const float n = tanhf   (gir[2*CFG_D + d] + r * ghr[2*CFG_D + d]);
    const float hv   = h[i];
    const float hnew = (1.0f - z) * n + z * hv;

    h[i] = hnew;
    const unsigned short hb = f2bf(hnew);
    h_bf[i] = hb;
    hs_t[i] = hb;
}

// ---------------------------------------------------------------------------
// Second output: caption_lengths - 1
// ---------------------------------------------------------------------------
__global__ void write_lengths(const long long* __restrict__ lens,
                              float* __restrict__ outf,
                              long long* __restrict__ outi, int as_i64)
{
    int b = threadIdx.x;
    if (b < CFG_B) {
        long long v = lens[b] - 1;
        if (as_i64) outi[b] = v;
        else        outf[b] = (float)v;
    }
}

// ---------------------------------------------------------------------------
// Host-side orchestration
// ---------------------------------------------------------------------------
static inline int cdiv(int a, int b) { return (a + b - 1) / b; }

extern "C" void kernel_launch(void* const* d_in, const int* in_sizes, int n_in,
                              void* d_out, int out_size, void* d_ws, size_t ws_size,
                              hipStream_t stream)
{
    (void)in_sizes; (void)n_in; (void)ws_size;

    const float*      gf     = (const float*)d_in[0];       // [128,2048]
    const long long*  caps   = (const long long*)d_in[1];   // [128,64]
    const long long*  lens   = (const long long*)d_in[2];   // [128]
    const float*      Wembed = (const float*)d_in[3];       // [10000,512]
    const float*      Winit  = (const float*)d_in[4];       // [512,2048]
    const float*      binit  = (const float*)d_in[5];       // [512]
    const float*      Wih    = (const float*)d_in[6];       // [1536,512]
    const float*      Whh    = (const float*)d_in[7];       // [1536,512]
    const float*      bih    = (const float*)d_in[8];       // [1536]
    const float*      bhh    = (const float*)d_in[9];       // [1536]
    const float*      Wfc    = (const float*)d_in[10];      // [10000,512]
    const float*      bfc    = (const float*)d_in[11];      // [10000]
    float*            out    = (float*)d_out;

    // ---- workspace bump allocator (256B aligned) ----
    char*  ws  = (char*)d_ws;
    size_t off = 0;
    auto alloc = [&](size_t bytes) -> void* {
        off = (off + 255) & ~(size_t)255;
        void* p = ws + off;
        off += bytes;
        return p;
    };
    unsigned short* gf_bf    = (unsigned short*)alloc((size_t)CFG_B  * CFG_ENC * 2);
    unsigned short* Winit_bf = (unsigned short*)alloc((size_t)CFG_D  * CFG_ENC * 2);
    unsigned short* Wih_bf   = (unsigned short*)alloc((size_t)CFG_G  * CFG_E   * 2);
    unsigned short* Whh_bf   = (unsigned short*)alloc((size_t)CFG_G  * CFG_D   * 2);
    unsigned short* Wfc_bf   = (unsigned short*)alloc((size_t)CFG_VP * CFG_D   * 2);
    unsigned short* emb_bf   = (unsigned short*)alloc((size_t)CFG_B  * CFG_L1 * CFG_E * 2);
    float*          gi       = (float*)alloc((size_t)CFG_L1 * CFG_B * CFG_G * 4);
    float*          gh       = (float*)alloc((size_t)CFG_B  * CFG_G * 4);
    float*          h        = (float*)alloc((size_t)CFG_B  * CFG_D * 4);
    unsigned short* h_bf     = (unsigned short*)alloc((size_t)CFG_B * CFG_D * 2);
    unsigned short* hs_bf    = (unsigned short*)alloc((size_t)CFG_L1 * CFG_B * CFG_D * 2);

    // ---- 1) convert operands to bf16 (W_fc zero-padded to 10112 rows) ----
    {
        struct { const float* src; unsigned short* dst; int rows; int cols; int rows_pad; }
        cv[5] = {
            { gf,    gf_bf,    CFG_B, CFG_ENC, CFG_B  },
            { Winit, Winit_bf, CFG_D, CFG_ENC, CFG_D  },
            { Wih,   Wih_bf,   CFG_G, CFG_E,   CFG_G  },
            { Whh,   Whh_bf,   CFG_G, CFG_D,   CFG_G  },
            { Wfc,   Wfc_bf,   CFG_V, CFG_D,   CFG_VP },
        };
        for (int i = 0; i < 5; ++i) {
            const long long total = (long long)cv[i].rows_pad * cv[i].cols;
            f32_to_bf16_pad<<<(int)((total + 255) / 256), 256, 0, stream>>>(
                cv[i].src, cv[i].dst, cv[i].rows, cv[i].cols, total);
        }
    }

    // ---- 2) embedding gather ----
    embed_gather<<<cdiv(CFG_B * CFG_L1 * CFG_E, 256), 256, 0, stream>>>(
        Wembed, caps, emb_bf);

    // ---- 3) h0 = tanh(gf @ Winit^T + b_init)  [128,512] ----
    gemm_bf16_wmma<<<dim3(CFG_D / 128, CFG_B / 64), 256, 0, stream>>>(
        (const __bf16*)gf_bf, (const __bf16*)Winit_bf, binit,
        h, h_bf, CFG_B, CFG_D, CFG_ENC, CFG_D, /*act=*/1, /*perm=*/0);

    // ---- 4) gi = emb @ Wih^T + b_ih, stored [l][b][1536] ----
    gemm_bf16_wmma<<<dim3(CFG_G / 128, (CFG_B * CFG_L1) / 64), 256, 0, stream>>>(
        (const __bf16*)emb_bf, (const __bf16*)Wih_bf, bih,
        gi, nullptr, CFG_B * CFG_L1, CFG_G, CFG_E, CFG_G, /*act=*/0, /*perm=*/1);

    // ---- 5) GRU scan: 63 steps of (gh GEMM + gate fuse) ----
    for (int t = 0; t < CFG_L1; ++t) {
        gemm_bf16_wmma<<<dim3(CFG_G / 128, CFG_B / 64), 256, 0, stream>>>(
            (const __bf16*)h_bf, (const __bf16*)Whh_bf, bhh,
            gh, nullptr, CFG_B, CFG_G, CFG_D, CFG_G, /*act=*/0, /*perm=*/0);
        gru_gate<<<cdiv(CFG_B * CFG_D, 256), 256, 0, stream>>>(
            gi + (size_t)t * CFG_B * CFG_G, gh,
            h, h_bf, hs_bf + (size_t)t * CFG_B * CFG_D);
    }

    // ---- 6) logits = hs @ Wfc^T + b_fc, stored [b][l][V] into d_out ----
    gemm_bf16_wmma<<<dim3(CFG_VP / 128, (CFG_L1 * CFG_B) / 64), 256, 0, stream>>>(
        (const __bf16*)hs_bf, (const __bf16*)Wfc_bf, bfc,
        out, nullptr, CFG_L1 * CFG_B, CFG_VP, CFG_D, CFG_V, /*act=*/0, /*perm=*/2);

    // ---- 7) caption_lengths - 1 appended after logits ----
    {
        const size_t logits_n = (size_t)CFG_B * CFG_L1 * CFG_V;
        const int as_i64 = ((size_t)out_size >= logits_n + 2 * CFG_B) ? 1 : 0;
        write_lengths<<<1, 128, 0, stream>>>(
            lens, out + logits_n, (long long*)(out + logits_n), as_i64);
    }
}
